// Seq2Seq_27736898797577
// MI455X (gfx1250) — matrix operand order repaired and checked by reading, not compiled, not run on hardware
//
#include <hip/hip_runtime.h>
#include <math.h>

// ---------------- model dims ----------------
#define kVOC  50000
#define kE    300
#define kEp   320        // E padded to multiple of 32
#define kHE   512
#define kHO   1024
#define kTN   5
#define kA    256
#define kL    100
#define kT    48
#define kB    32
#define kDIN  1624       // E + 2*HE + E
#define kDINp 1632       // padded to multiple of 32
#define kHD   1024

typedef __attribute__((ext_vector_type(16))) __bf16 v16bf;
typedef __attribute__((ext_vector_type(8)))  float  v8f;

union V16U { uint4 q[2]; v16bf v; };

// ---- CDNA5 16-bit A-matrix 16x32 fragment (ISA 7.12.2), row-major source ----
// lanes 0-15 : m = lane,   K chunks {0..7, 16..23}
// lanes16-31 : m = lane-16, K chunks {8..15, 24..31}
__device__ __forceinline__ v16bf load_A_frag(const __bf16* A, int lda,
                                             int mBase, int kBase, int lane) {
  int m = mBase + (lane & 15);
  int kHalf = ((lane >> 4) & 1) * 8;
  const __bf16* p = A + (size_t)m * lda + kBase + kHalf;
  V16U u;
  u.q[0] = *reinterpret_cast<const uint4*>(p);        // K = kHalf+0..7
  u.q[1] = *reinterpret_cast<const uint4*>(p + 16);   // K = kHalf+16..23
  return u.v;
}

// ---- B-matrix fragment from PRE-PACKED layout ----
// Weights are packed by pack_b into [nTile][kTile][lane][16 elems] so that a
// wave's fragment load is one contiguous, fully-coalesced 1KB burst.
__device__ __forceinline__ v16bf load_B_packed(const __bf16* Bp, int Ktiles,
                                               int nTile, int kTile, int lane) {
  const __bf16* p = Bp + ((((size_t)nTile * Ktiles + kTile) * 32 + lane) << 4);
  V16U u;
  u.q[0] = *reinterpret_cast<const uint4*>(p);
  u.q[1] = *reinterpret_cast<const uint4*>(p + 8);
  return u.v;
}

// ---------------------------------------------------------------------------
// Generic WMMA GEMM:  C[M,N](f32) = A0(M,K0)bf16 @ B0packed
//                                  [+ A1(M,K1)bf16 @ B1packed] + bias
// grid.x: N in 128-col blocks (8 waves * 16); grid.y: M in 32s.
// K0/K1 are padded strides (multiple of 32, zero padded).
// ---------------------------------------------------------------------------
__global__ __launch_bounds__(256) void wmma_gemm(
    const __bf16* __restrict__ A0, const __bf16* __restrict__ B0, int K0,
    const __bf16* __restrict__ A1, const __bf16* __restrict__ B1, int K1,
    const float* __restrict__ bias, float* __restrict__ C,
    int ldc, int N, int Nalloc) {
  int lane = threadIdx.x & 31;
  int wave = threadIdx.x >> 5;
  int nBase = blockIdx.x * 128 + wave * 16;
  if (nBase >= Nalloc) return;            // wave-uniform, no barriers in kernel
  int mBase = blockIdx.y * 32;
  int nT = nBase >> 4;

  v8f acc0 = {};
  v8f acc1 = {};
  for (int k = 0; k < K0; k += 32) {
    v16bf bf = load_B_packed(B0, K0 >> 5, nT, k >> 5, lane);
    v16bf a0 = load_A_frag(A0, K0, mBase,      k, lane);
    v16bf a1 = load_A_frag(A0, K0, mBase + 16, k, lane);
    acc0 = __builtin_amdgcn_wmma_f32_16x16x32_bf16(false, a0, false, bf,
                                                   (short)0, acc0, false, false);
    acc1 = __builtin_amdgcn_wmma_f32_16x16x32_bf16(false, a1, false, bf,
                                                   (short)0, acc1, false, false);
  }
  if (A1 != nullptr) {
    for (int k = 0; k < K1; k += 32) {
      v16bf bf = load_B_packed(B1, K1 >> 5, nT, k >> 5, lane);
      v16bf a0 = load_A_frag(A1, K1, mBase,      k, lane);
      v16bf a1 = load_A_frag(A1, K1, mBase + 16, k, lane);
      acc0 = __builtin_amdgcn_wmma_f32_16x16x32_bf16(false, a0, false, bf,
                                                     (short)0, acc0, false, false);
      acc1 = __builtin_amdgcn_wmma_f32_16x16x32_bf16(false, a1, false, bf,
                                                     (short)0, acc1, false, false);
    }
  }
  // C/D layout: VGPR r -> M = r (lanes 0-15) or 8+r (lanes 16-31); N = lane%16
  int n = nBase + (lane & 15);
  bool nOK = (n < N);
  float bv = (bias != nullptr && nOK) ? bias[n] : 0.f;
  int mOff = (lane >> 4) ? 8 : 0;
#pragma unroll
  for (int r = 0; r < 8; ++r) {
    if (nOK) {
      C[(size_t)(mBase + mOff + r) * ldc + n]      = acc0[r] + bv;
      C[(size_t)(mBase + 16 + mOff + r) * ldc + n] = acc1[r] + bv;
    }
  }
}

// ---------------------------------------------------------------------------
// Wide WMMA GEMM for the dominant logits matmul: M=32, each wave owns a
// 32x32 output tile (2 M-tiles x 2 N-tiles) -> 2 coalesced loads per WMMA.
// Epilogue uses NON-TEMPORAL stores so the 307MB logits stream does not evict
// the L2-resident bf16 Wfc (102.4MB of 192MB L2).
// ---------------------------------------------------------------------------
__global__ __launch_bounds__(256) void wmma_gemm_wide(
    const __bf16* __restrict__ A0, const __bf16* __restrict__ B0, int K0,
    const float* __restrict__ bias, float* __restrict__ C,
    int ldc, int N, int Nalloc) {
  int lane = threadIdx.x & 31;
  int wave = threadIdx.x >> 5;
  int nBase = blockIdx.x * 256 + wave * 32;
  if (nBase >= Nalloc) return;
  // clamp second tile (duplicate-compute of a valid tile; identical values)
  int n1Base = nBase + 16;
  if (n1Base + 16 > Nalloc) n1Base = Nalloc - 16;
  int nT0 = nBase >> 4, nT1 = n1Base >> 4;
  int Ktiles = K0 >> 5;

  v8f acc00 = {}, acc01 = {}, acc10 = {}, acc11 = {};
  for (int k = 0; k < K0; k += 32) {
    int ktl = k >> 5;
    v16bf b0 = load_B_packed(B0, Ktiles, nT0, ktl, lane);
    v16bf b1 = load_B_packed(B0, Ktiles, nT1, ktl, lane);
    v16bf a0 = load_A_frag(A0, K0, 0,  k, lane);
    v16bf a1 = load_A_frag(A0, K0, 16, k, lane);
    acc00 = __builtin_amdgcn_wmma_f32_16x16x32_bf16(false, a0, false, b0,
                                                    (short)0, acc00, false, false);
    acc10 = __builtin_amdgcn_wmma_f32_16x16x32_bf16(false, a1, false, b0,
                                                    (short)0, acc10, false, false);
    acc01 = __builtin_amdgcn_wmma_f32_16x16x32_bf16(false, a0, false, b1,
                                                    (short)0, acc01, false, false);
    acc11 = __builtin_amdgcn_wmma_f32_16x16x32_bf16(false, a1, false, b1,
                                                    (short)0, acc11, false, false);
  }
  int nl = lane & 15;
  int n0 = nBase + nl, n1 = n1Base + nl;
  bool ok0 = (n0 < N), ok1 = (n1 < N);
  float bv0 = (bias && ok0) ? bias[n0] : 0.f;
  float bv1 = (bias && ok1) ? bias[n1] : 0.f;
  int mOff = (lane >> 4) ? 8 : 0;
#pragma unroll
  for (int r = 0; r < 8; ++r) {
    size_t row0 = (size_t)(mOff + r) * ldc;
    size_t row1 = (size_t)(16 + mOff + r) * ldc;
    if (ok0) {
      __builtin_nontemporal_store(acc00[r] + bv0, &C[row0 + n0]);
      __builtin_nontemporal_store(acc10[r] + bv0, &C[row1 + n0]);
    }
    if (ok1) {
      __builtin_nontemporal_store(acc01[r] + bv1, &C[row0 + n1]);
      __builtin_nontemporal_store(acc11[r] + bv1, &C[row1 + n1]);
    }
  }
}

// ---------------- helpers / pointwise kernels ----------------
__device__ __forceinline__ float sigm(float x) { return 1.f / (1.f + __expf(-x)); }

// Convert fp32 weight (rows x cols) into packed bf16 B-fragment layout
// [nTile][kTile][lane][16 elems], zero-padded to rowsPad x colsPad.
// NT loads: the fp32 master weights are read exactly once -> don't cache them.
__global__ void pack_b(const float* __restrict__ src, __bf16* __restrict__ dst,
                       int rows, int cols, int rowsPad, int colsPad) {
  int i = blockIdx.x * blockDim.x + threadIdx.x;
  if (i >= rowsPad * colsPad) return;
  int e = i & 15;
  int lane = (i >> 4) & 31;
  int t = i >> 9;
  int Ktiles = colsPad >> 5;
  int ktl = t % Ktiles;
  int nT = t / Ktiles;
  int n = nT * 16 + (lane & 15);
  int k = ktl * 32 + ((lane >> 4) & 1) * 8 + (e & 7) + (e >> 3) * 16;
  float v = 0.f;
  if (n < rows && k < cols)
    v = __builtin_nontemporal_load(&src[(size_t)n * cols + k]);
  dst[i] = (__bf16)v;
}

__global__ void cvt_pad(const float* __restrict__ src, __bf16* __restrict__ dst,
                        int rows, int cols, int rowsPad, int colsPad) {
  int i = blockIdx.x * blockDim.x + threadIdx.x;
  if (i >= rowsPad * colsPad) return;
  int r = i / colsPad, c = i % colsPad;
  float v = (r < rows && c < cols) ? src[(size_t)r * cols + c] : 0.f;
  dst[i] = (__bf16)v;
}

__global__ void zero_f32(float* p, int n) {
  int i = blockIdx.x * blockDim.x + threadIdx.x;
  if (i < n) p[i] = 0.f;
}
__global__ void zero_bf16(__bf16* p, int n) {
  int i = blockIdx.x * blockDim.x + threadIdx.x;
  if (i < n) p[i] = (__bf16)0.f;
}

// x[t,b,:] = enc_emb[topics[b,t]]  (padded bf16, (TN*B, Ep))
__global__ void gather_topics(const float* __restrict__ tab,
                              const int* __restrict__ topics, __bf16* __restrict__ x) {
  int i = blockIdx.x * blockDim.x + threadIdx.x;
  if (i >= kTN * kB * kEp) return;
  int c = i % kEp, r = i / kEp;         // r = t*32 + b
  int tt = r / kB, b = r % kB;
  int tok = topics[b * kTN + tt];
  float v = (c < kE) ? tab[(size_t)tok * kE + c] : 0.f;
  x[i] = (__bf16)v;
}

// mem0 = mem_emb[mems] -> fp32 (B*L, E) and padded bf16 (B*L, Ep)
__global__ void gather_mem(const float* __restrict__ tab, const int* __restrict__ mems,
                           __bf16* __restrict__ dstB, float* __restrict__ dstF) {
  int i = blockIdx.x * blockDim.x + threadIdx.x;
  if (i >= kB * kL * kEp) return;
  int r = i / kEp, c = i % kEp;
  int tok = mems[r];
  float v = (c < kE) ? tab[(size_t)tok * kE + c] : 0.f;
  dstB[i] = (__bf16)v;
  if (c < kE) dstF[(size_t)r * kE + c] = v;
}

// emb = dec_emb[essay[:, clamp(t+1)]]  (t = -1 gives column 0)
__global__ void gather_tok(const float* __restrict__ tab, const int* __restrict__ essay,
                           int t, __bf16* __restrict__ embB, float* __restrict__ embF) {
  int i = blockIdx.x * blockDim.x + threadIdx.x;
  if (i >= kB * kEp) return;
  int b = i / kEp, c = i % kEp;
  int col = t + 1;
  if (col < 0) col = 0;
  if (col > kT - 1) col = kT - 1;
  int tok = essay[b * kT + col];
  float v = (c < kE) ? tab[(size_t)tok * kE + c] : 0.f;
  embB[i] = (__bf16)v;
  if (c < kE) embF[b * kE + c] = v;
}

// LSTM nonlinearity; gates (B,4H) pre-bias; writes h,c (f32), h,c (bf16),
// optional ys slice (stride ysLd).
__global__ void lstm_pw(const float* __restrict__ gates, const float* __restrict__ bias,
                        const float* __restrict__ cPrev, float* __restrict__ hOut,
                        float* __restrict__ cOut, __bf16* __restrict__ hB,
                        __bf16* __restrict__ cB, int H,
                        float* __restrict__ ysOut, int ysLd) {
  int i = blockIdx.x * blockDim.x + threadIdx.x;
  if (i >= kB * H) return;
  int b = i / H, j = i % H;
  const float* g = gates + (size_t)b * 4 * H;
  float gi = g[j] + bias[j];
  float gf = g[H + j] + bias[H + j];
  float gg = g[2 * H + j] + bias[2 * H + j];
  float go = g[3 * H + j] + bias[3 * H + j];
  float c = sigm(gf) * cPrev[i] + sigm(gi) * tanhf(gg);
  float h = sigm(go) * tanhf(c);
  cOut[i] = c;
  hOut[i] = h;
  hB[i] = (__bf16)h;
  if (cB) cB[i] = (__bf16)c;
  if (ysOut) ysOut[(size_t)b * ysLd + j] = h;
}

// h0 = stack([hf,hb]).reshape(B, 2HE)  (reference's raw-flat reinterpretation)
__global__ void build_h0c0(const float* __restrict__ hf, const float* __restrict__ hb,
                           const float* __restrict__ cf, const float* __restrict__ cb,
                           float* __restrict__ h0, float* __restrict__ c0,
                           __bf16* __restrict__ hB, __bf16* __restrict__ cB) {
  int i = blockIdx.x * blockDim.x + threadIdx.x;
  if (i >= kB * 2 * kHE) return;
  int s = i >> 14, b = (i >> 9) & 31, k = i & 511;
  float h = (s ? hb : hf)[b * kHE + k];
  float c = (s ? cb : cf)[b * kHE + k];
  h0[i] = h;
  c0[i] = c;
  hB[i] = (__bf16)h;
  cB[i] = (__bf16)c;
}

// topic attention: e = tanh(pre@q) @ W3^T + b3 ; alpha = softmax ; c_t (B,1024)
__global__ void attn_kernel(const float* __restrict__ pre, const float* __restrict__ q,
                            const float* __restrict__ W3, const float* __restrict__ b3,
                            const float* __restrict__ TR, float* __restrict__ cT) {
  int b = blockIdx.x;
  __shared__ float red[256];
  __shared__ float s5[kTN], e5[kTN], al[kTN];
  float qv = q[b * kA + threadIdx.x];
  for (int t = 0; t < kTN; ++t) {
    red[threadIdx.x] = pre[(size_t)(b * kTN + t) * kA + threadIdx.x] * qv;
    __syncthreads();
    for (int off = 128; off; off >>= 1) {
      if ((int)threadIdx.x < off) red[threadIdx.x] += red[threadIdx.x + off];
      __syncthreads();
    }
    if (threadIdx.x == 0) s5[t] = tanhf(red[0]);
    __syncthreads();
  }
  if (threadIdx.x < kTN) {
    float e = b3[threadIdx.x];
    for (int t = 0; t < kTN; ++t) e += s5[t] * W3[threadIdx.x * kTN + t];
    e5[threadIdx.x] = e;
  }
  __syncthreads();
  if (threadIdx.x == 0) {
    float mx = e5[0];
    for (int t = 1; t < kTN; ++t) mx = fmaxf(mx, e5[t]);
    float sm = 0.f;
    for (int t = 0; t < kTN; ++t) { al[t] = __expf(e5[t] - mx); sm += al[t]; }
    for (int t = 0; t < kTN; ++t) al[t] /= sm;
  }
  __syncthreads();
  // c_t[b,j] = sum_t TRflat[b*5120 + j*5 + t] * alpha[t]   (reshape(B,-1,TN))
  for (int j = threadIdx.x; j < 2 * kHE; j += 256) {
    const float* base = TR + (size_t)b * (2 * kHE * kTN) + (size_t)j * kTN;
    float v = 0.f;
    for (int t = 0; t < kTN; ++t) v += base[t] * al[t];
    cT[b * (2 * kHE) + j] = v;
  }
}

// v = tanh(vraw); qm = softmax(mem @ v); m = qm^T @ mem
__global__ void memv_kernel(const float* __restrict__ vraw, int ldv,
                            const float* __restrict__ mem, float* __restrict__ mOut) {
  int b = blockIdx.x;
  __shared__ float v[kE];
  __shared__ float sc[kL], al[kL];
  for (int j = threadIdx.x; j < kE; j += 256) v[j] = tanhf(vraw[b * ldv + j]);
  __syncthreads();
  for (int l = threadIdx.x; l < kL; l += 256) {
    const float* mr = mem + ((size_t)b * kL + l) * kE;
    float s = 0.f;
    for (int k = 0; k < kE; ++k) s += mr[k] * v[k];
    sc[l] = s;
  }
  __syncthreads();
  if (threadIdx.x == 0) {
    float mx = sc[0];
    for (int l = 1; l < kL; ++l) mx = fmaxf(mx, sc[l]);
    float sm = 0.f;
    for (int l = 0; l < kL; ++l) { al[l] = __expf(sc[l] - mx); sm += al[l]; }
    float inv = 1.f / sm;
    for (int l = 0; l < kL; ++l) al[l] *= inv;
  }
  __syncthreads();
  for (int k = threadIdx.x; k < kE; k += 256) {
    float s = 0.f;
    for (int l = 0; l < kL; ++l) s += al[l] * mem[((size_t)b * kL + l) * kE + k];
    mOut[b * kE + k] = s;
  }
}

// mem2 = tanh(mU1+eV1)*sigm(mU2+eV2) + mem*(1-sigm(...)), biases folded in GEMM
__global__ void update_mem(const float* __restrict__ mU1, const float* __restrict__ mU2,
                           int ldu, const float* __restrict__ eV1,
                           const float* __restrict__ eV2, int ldv,
                           float* __restrict__ mem, __bf16* __restrict__ memB) {
  int i = blockIdx.x * blockDim.x + threadIdx.x;
  if (i >= kB * kL * kE) return;
  int r = i / kE, j = i % kE;
  int b = r / kL;
  float Mt = tanhf(mU1[(size_t)r * ldu + j] + eV1[(size_t)b * ldv + j]);
  float g  = sigm(mU2[(size_t)r * ldu + j] + eV2[(size_t)b * ldv + j]);
  float m2 = Mt * g + mem[i] * (1.f - g);
  mem[i] = m2;
  memB[(size_t)r * kEp + j] = (__bf16)m2;   // pad cols stay zero from gather
}

// dec_in = concat(emb, c_t, m) -> padded bf16 (B, DINp)
__global__ void build_dec_in(const float* __restrict__ emb, const float* __restrict__ cT,
                             const float* __restrict__ m, __bf16* __restrict__ decIn) {
  int i = blockIdx.x * blockDim.x + threadIdx.x;
  if (i >= kB * kDINp) return;
  int b = i / kDINp, j = i % kDINp;
  float v;
  if (j < kE)            v = emb[b * kE + j];
  else if (j < kE + kHO) v = cT[b * kHO + (j - kE)];
  else if (j < kDIN)     v = m[b * kE + (j - kE - kHO)];
  else                   v = 0.f;
  decIn[i] = (__bf16)v;
}

// ---------------------------------------------------------------------------
extern "C" void kernel_launch(void* const* d_in, const int* in_sizes, int n_in,
                              void* d_out, int out_size, void* d_ws, size_t ws_size,
                              hipStream_t stream) {
  (void)in_sizes; (void)n_in; (void)out_size; (void)ws_size;
  const int*   topics   = (const int*)d_in[0];
  const int*   essay    = (const int*)d_in[1];
  const int*   mems     = (const int*)d_in[2];
  const float* enc_emb  = (const float*)d_in[3];
  const float* eWihF    = (const float*)d_in[4];
  const float* eWhhF    = (const float*)d_in[5];
  const float* eBF      = (const float*)d_in[6];
  const float* eWihB    = (const float*)d_in[7];
  const float* eWhhB    = (const float*)d_in[8];
  const float* eBB      = (const float*)d_in[9];
  const float* dec_emb  = (const float*)d_in[10];
  const float* dWih     = (const float*)d_in[11];
  const float* dWhh     = (const float*)d_in[12];
  const float* dB       = (const float*)d_in[13];
  const float* Wfc      = (const float*)d_in[14];
  const float* bfc      = (const float*)d_in[15];
  const float* mem_emb  = (const float*)d_in[16];
  const float* Wm       = (const float*)d_in[17];
  const float* bm       = (const float*)d_in[18];
  const float* U1       = (const float*)d_in[19];
  const float* bU1      = (const float*)d_in[20];
  const float* V1       = (const float*)d_in[21];
  const float* bV1      = (const float*)d_in[22];
  const float* U2       = (const float*)d_in[23];
  const float* bU2      = (const float*)d_in[24];
  const float* V2       = (const float*)d_in[25];
  const float* bV2      = (const float*)d_in[26];
  const float* W1       = (const float*)d_in[27];
  const float* b1       = (const float*)d_in[28];
  const float* W2       = (const float*)d_in[29];
  const float* b2       = (const float*)d_in[30];
  const float* W3       = (const float*)d_in[31];
  const float* b3       = (const float*)d_in[32];
  float* out = (float*)d_out;

  // ---- bump allocator over d_ws ----
  char* ws = (char*)d_ws;
  size_t off = 0;
  auto alloc = [&](size_t bytes) -> void* {
    void* p = ws + off;
    off = (off + bytes + 255) & ~(size_t)255;
    return p;
  };
  // packed bf16 weights (fragment-order, zero padded)
  __bf16* eWihFb = (__bf16*)alloc((size_t)4 * kHE * kEp * 2);
  __bf16* eWhhFb = (__bf16*)alloc((size_t)4 * kHE * kHE * 2);
  __bf16* eWihBb = (__bf16*)alloc((size_t)4 * kHE * kEp * 2);
  __bf16* eWhhBb = (__bf16*)alloc((size_t)4 * kHE * kHE * 2);
  __bf16* dWihb  = (__bf16*)alloc((size_t)4 * kHD * kDINp * 2);
  __bf16* dWhhb  = (__bf16*)alloc((size_t)4 * kHD * kHD * 2);
  __bf16* Wfcb   = (__bf16*)alloc((size_t)kVOC * kHD * 2);
  __bf16* Wmb    = (__bf16*)alloc((size_t)kEp * kHD * 2);
  __bf16* U1b    = (__bf16*)alloc((size_t)kEp * kEp * 2);
  __bf16* V1b    = (__bf16*)alloc((size_t)kEp * kEp * 2);
  __bf16* U2b    = (__bf16*)alloc((size_t)kEp * kEp * 2);
  __bf16* V2b    = (__bf16*)alloc((size_t)kEp * kEp * 2);
  __bf16* W1b    = (__bf16*)alloc((size_t)kA * (2 * kHE) * 2);
  __bf16* W2b    = (__bf16*)alloc((size_t)kA * kHD * 2);
  // activations
  __bf16* xEnc   = (__bf16*)alloc((size_t)kTN * kB * kEp * 2);
  float*  TR     = (float*)alloc((size_t)kTN * kB * 2 * kHE * 4);
  __bf16* TRb    = (__bf16*)alloc((size_t)kTN * kB * 2 * kHE * 2);
  float*  preBuf = (float*)alloc((size_t)kB * kTN * kA * 4);
  float*  encSt  = (float*)alloc((size_t)4 * kB * kHE * 4);   // hf,cf,hb,cb
  float *hf = encSt, *cf = encSt + kB * kHE, *hb = encSt + 2 * kB * kHE,
        *cb = encSt + 3 * kB * kHE;
  __bf16* encHB  = (__bf16*)alloc((size_t)2 * kB * kHE * 2);  // hfB,hbB
  __bf16 *hfB = encHB, *hbB = encHB + kB * kHE;
  float*  gatesE = (float*)alloc((size_t)kB * 4 * kHE * 4);
  float*  hF     = (float*)alloc((size_t)kB * kHD * 4);
  float*  cF     = (float*)alloc((size_t)kB * kHD * 4);
  __bf16* hB     = (__bf16*)alloc((size_t)kB * kHD * 2);
  __bf16* cB     = (__bf16*)alloc((size_t)kB * kHD * 2);
  float*  gatesD = (float*)alloc((size_t)kB * 4 * kHD * 4);
  __bf16* decIn  = (__bf16*)alloc((size_t)kB * kDINp * 2);
  float*  memF   = (float*)alloc((size_t)kB * kL * kE * 4);
  __bf16* memB   = (__bf16*)alloc((size_t)kB * kL * kEp * 2);
  float*  mU1    = (float*)alloc((size_t)kB * kL * kEp * 4);
  float*  mU2    = (float*)alloc((size_t)kB * kL * kEp * 4);
  float*  eV1    = (float*)alloc((size_t)kB * kEp * 4);
  float*  eV2    = (float*)alloc((size_t)kB * kEp * 4);
  float*  qBuf   = (float*)alloc((size_t)kB * kA * 4);
  float*  cT     = (float*)alloc((size_t)kB * kHO * 4);
  float*  vraw   = (float*)alloc((size_t)kB * kEp * 4);
  float*  mBuf   = (float*)alloc((size_t)kB * kE * 4);
  float*  embF   = (float*)alloc((size_t)kB * kE * 4);
  __bf16* embB   = (__bf16*)alloc((size_t)kB * kEp * 2);

  auto pack = [&](const float* s, __bf16* d, int r, int c, int rp, int cp) {
    int total = rp * cp;
    pack_b<<<(total + 255) / 256, 256, 0, stream>>>(s, d, r, c, rp, cp);
  };
  auto gemm = [&](const __bf16* A0, const __bf16* B0, int K0, const __bf16* A1,
                  const __bf16* B1, int K1, const float* bias, float* C, int ldc,
                  int N, int Nalloc, int M) {
    dim3 grid((N + 127) / 128, M / 32);
    wmma_gemm<<<grid, 256, 0, stream>>>(A0, B0, K0, A1, B1, K1, bias, C, ldc, N,
                                        Nalloc);
  };

  // ---- per-launch weight conversion into packed, L2-resident bf16 ----
  pack(eWihF, eWihFb, 4 * kHE, kE, 4 * kHE, kEp);
  pack(eWhhF, eWhhFb, 4 * kHE, kHE, 4 * kHE, kHE);
  pack(eWihB, eWihBb, 4 * kHE, kE, 4 * kHE, kEp);
  pack(eWhhB, eWhhBb, 4 * kHE, kHE, 4 * kHE, kHE);
  pack(dWih, dWihb, 4 * kHD, kDIN, 4 * kHD, kDINp);
  pack(dWhh, dWhhb, 4 * kHD, kHD, 4 * kHD, kHD);
  pack(Wfc, Wfcb, kVOC, kHD, kVOC, kHD);    // 102.4MB bf16 -> L2-resident
  pack(Wm, Wmb, kE, kHD, kEp, kHD);
  pack(U1, U1b, kE, kE, kEp, kEp);
  pack(V1, V1b, kE, kE, kEp, kEp);
  pack(U2, U2b, kE, kE, kEp, kEp);
  pack(V2, V2b, kE, kE, kEp, kEp);
  pack(W1, W1b, kA, 2 * kHE, kA, 2 * kHE);
  pack(W2, W2b, kA, kHD, kA, kHD);

  // ---- gathers + encoder state init ----
  gather_topics<<<(kTN * kB * kEp + 255) / 256, 256, 0, stream>>>(enc_emb, topics,
                                                                  xEnc);
  gather_mem<<<(kB * kL * kEp + 255) / 256, 256, 0, stream>>>(mem_emb, mems, memB,
                                                              memF);
  zero_f32<<<(4 * kB * kHE + 255) / 256, 256, 0, stream>>>(encSt, 4 * kB * kHE);
  zero_bf16<<<(2 * kB * kHE + 255) / 256, 256, 0, stream>>>(encHB, 2 * kB * kHE);

  // ---- bidirectional encoder (TN steps) ----
  for (int t = 0; t < kTN; ++t) {
    gemm(xEnc + (size_t)t * kB * kEp, eWihFb, kEp, hfB, eWhhFb, kHE, nullptr,
         gatesE, 4 * kHE, 4 * kHE, 4 * kHE, kB);
    lstm_pw<<<(kB * kHE + 255) / 256, 256, 0, stream>>>(
        gatesE, eBF, cf, hf, cf, hfB, (__bf16*)nullptr, kHE,
        TR + (size_t)t * kB * 2 * kHE, 2 * kHE);
    int tb = kTN - 1 - t;
    gemm(xEnc + (size_t)tb * kB * kEp, eWihBb, kEp, hbB, eWhhBb, kHE, nullptr,
         gatesE, 4 * kHE, 4 * kHE, 4 * kHE, kB);
    lstm_pw<<<(kB * kHE + 255) / 256, 256, 0, stream>>>(
        gatesE, eBB, cb, hb, cb, hbB, (__bf16*)nullptr, kHE,
        TR + (size_t)tb * kB * 2 * kHE + kHE, 2 * kHE);
  }
  build_h0c0<<<(kB * 2 * kHE + 255) / 256, 256, 0, stream>>>(hf, hb, cf, cb, hF, cF,
                                                             hB, cB);
  // loop-invariant: pre = topics_repr.reshape(160,1024) @ W1^T + b1
  cvt_pad<<<(kTN * kB * 2 * kHE + 255) / 256, 256, 0, stream>>>(
      TR, TRb, kTN * kB, 2 * kHE, kTN * kB, 2 * kHE);
  gemm(TRb, W1b, 2 * kHE, nullptr, nullptr, 0, b1, preBuf, kA, kA, kA, kTN * kB);

  auto run_before_feed = [&]() {
    gemm(cB, W2b, kHD, nullptr, nullptr, 0, b2, qBuf, kA, kA, kA, kB);
    attn_kernel<<<kB, 256, 0, stream>>>(preBuf, qBuf, W3, b3, TR, cT);
    gemm(hB, Wmb, kHD, nullptr, nullptr, 0, bm, vraw, kEp, kE, kEp, kB);
    memv_kernel<<<kB, 256, 0, stream>>>(vraw, kEp, memF, mBuf);
    build_dec_in<<<(kB * kDINp + 255) / 256, 256, 0, stream>>>(embF, cT, mBuf,
                                                               decIn);
  };
  auto run_update_mem = [&]() {
    gemm(memB, U1b, kEp, nullptr, nullptr, 0, bU1, mU1, kEp, kE, kEp, kB * kL);
    gemm(memB, U2b, kEp, nullptr, nullptr, 0, bU2, mU2, kEp, kE, kEp, kB * kL);
    gemm(embB, V1b, kEp, nullptr, nullptr, 0, bV1, eV1, kEp, kE, kEp, kB);
    gemm(embB, V2b, kEp, nullptr, nullptr, 0, bV2, eV2, kEp, kE, kEp, kB);
    update_mem<<<(kB * kL * kE + 255) / 256, 256, 0, stream>>>(mU1, mU2, kEp, eV1,
                                                               eV2, kEp, memF, memB);
  };

  // initial before_feed with emb0 = dec_emb[essay[:,0]], h0/c0, mem0
  gather_tok<<<(kB * kEp + 255) / 256, 256, 0, stream>>>(dec_emb, essay, -1, embB,
                                                         embF);
  run_before_feed();

  // ---- decoder: T sequential steps ----
  for (int t = 0; t < kT; ++t) {
    gemm(decIn, dWihb, kDINp, hB, dWhhb, kHD, nullptr, gatesD, 4 * kHD, 4 * kHD,
         4 * kHD, kB);
    lstm_pw<<<(kB * kHD + 255) / 256, 256, 0, stream>>>(gatesD, dB, cF, hF, cF, hB,
                                                        cB, kHD, (float*)nullptr, 0);
    // dominant GEMM: logits -> d_out[t], wide tiles + NT stores
    wmma_gemm_wide<<<dim3((kVOC + 255) / 256, 1), 256, 0, stream>>>(
        hB, Wfcb, kHD, bfc, out + (size_t)t * kB * kVOC, kVOC, kVOC, kVOC);
    gather_tok<<<(kB * kEp + 255) / 256, 256, 0, stream>>>(dec_emb, essay, t, embB,
                                                           embF);
    run_update_mem();
    run_before_feed();
  }
}
